// DroplessMoE_17626545783344
// MI455X (gfx1250) — matrix operand order repaired and checked by reading, not compiled, not run on hardware
//
#include <hip/hip_runtime.h>
#include <math.h>

typedef __attribute__((ext_vector_type(16))) _Float16 v16h;
typedef __attribute__((ext_vector_type(8)))  float    v8f;
typedef int v4i __attribute__((vector_size(16)));

#define TOKENS 4096   // B*S
#define HDIM   1024
#define FDIM   4096
#define NEXP   8

#define TILE_M   48                 // tokens per FFN block
#define MSUB     (TILE_M / 16)      // 3 x 16-row WMMA subtiles
#define FCHUNK   256                // F columns per chunk (16 per wave, 16 waves)
#define HCHUNK   128                // x staging chunk along H
#define NT_FFN   512                // 16 waves
#define NTHREADS 256                // router/scatter blocks

#define GLOBAL_AS __attribute__((address_space(1)))
#define LDS_AS    __attribute__((address_space(3)))

#if defined(__has_builtin)
#if __has_builtin(__builtin_amdgcn_global_load_async_to_lds_b128)
#define USE_ASYNC_LDS 1
#endif
#endif

// ---------------------------------------------------------------------------
// WMMA fragment index helpers (16-bit A-matrix 16x32 layout, ISA 05_wmma §7.12.2)
// lane<16: M=lane, K={0..7,16..23}; lane>=16: M=lane-16, K={8..15,24..31}
// ---------------------------------------------------------------------------
__device__ __forceinline__ int a_frag_k(int i, int hi) {
    int vg = i >> 1;
    return ((vg >= 4) ? 16 : 0) + hi * 8 + (vg & 3) * 2 + (i & 1);
}
__device__ __forceinline__ void a_frag_inv(int M, int kk, int& lane, int& i) {
    int hi  = (kk >> 3) & 1;
    int vg  = ((kk >> 4) & 1) * 4 + ((kk >> 1) & 3);
    int par = kk & 1;
    lane = hi * 16 + M;
    i = vg * 2 + par;
}

// ---------------------------------------------------------------------------
// Kernel 0: zero output + small accumulators (we atomicAdd into d_out)
// ---------------------------------------------------------------------------
__global__ void zero_kernel(float* __restrict__ out, int n_out,
                            float* __restrict__ acc5, int* __restrict__ counts) {
    int i = blockIdx.x * blockDim.x + threadIdx.x;
    for (int k = i; k < n_out; k += gridDim.x * blockDim.x) out[k] = 0.0f;
    if (i < 8) acc5[i] = 0.0f;
    if (i < NEXP) counts[i] = 0;
}

// ---------------------------------------------------------------------------
// Kernel 1: router — logits, softmax, top-2, loss accumulators, expert counts
// ---------------------------------------------------------------------------
__global__ __launch_bounds__(NTHREADS)
void router_kernel(const float* __restrict__ x, const float* __restrict__ gw,
                   float* __restrict__ rw, int* __restrict__ idx,
                   int* __restrict__ counts, float* __restrict__ acc5) {
    __shared__ float sg[NEXP][HDIM];                 // 32 KB gate weights
    int tid = threadIdx.x;
    for (int i = tid; i < NEXP * HDIM; i += NTHREADS)
        sg[i >> 10][i & (HDIM - 1)] = gw[i];
    __syncthreads();

    int t = blockIdx.x * NTHREADS + tid;
    const float* xr = &x[(size_t)t * HDIM];
    float logit[NEXP];
#pragma unroll
    for (int e = 0; e < NEXP; ++e) logit[e] = 0.0f;
    for (int h = 0; h < HDIM; ++h) {
        float xv = xr[h];
#pragma unroll
        for (int e = 0; e < NEXP; ++e) logit[e] = fmaf(xv, sg[e][h], logit[e]);
    }
    float m = logit[0];
#pragma unroll
    for (int e = 1; e < NEXP; ++e) m = fmaxf(m, logit[e]);
    float p[NEXP], sum = 0.0f;
#pragma unroll
    for (int e = 0; e < NEXP; ++e) { p[e] = __expf(logit[e] - m); sum += p[e]; }
    float inv = 1.0f / sum;
#pragma unroll
    for (int e = 0; e < NEXP; ++e) p[e] *= inv;
    // top-2 (ties -> lowest index, matching jax top_k)
    int e0 = 0; float p0 = p[0];
#pragma unroll
    for (int e = 1; e < NEXP; ++e) if (p[e] > p0) { p0 = p[e]; e0 = e; }
    int e1 = (e0 == 0) ? 1 : 0; float p1 = p[e1];
#pragma unroll
    for (int e = 0; e < NEXP; ++e)
        if (e != e0 && p[e] > p1) { p1 = p[e]; e1 = e; }

    float wsum = p0 + p1;
    rw[t * 2 + 0] = p0 / wsum;  rw[t * 2 + 1] = p1 / wsum;
    idx[t * 2 + 0] = e0;        idx[t * 2 + 1] = e1;
    atomicAdd(&counts[e0], 1);  atomicAdd(&counts[e1], 1);

    float log_z = m + __logf(sum);
    // aux-loss literal one_hot(num_classes=2): classes 0 and 1 only
    float m0 = (e0 == 0 || e1 == 0) ? 1.0f : 0.0f;
    float m1 = (e0 == 1 || e1 == 1) ? 1.0f : 0.0f;
    atomicAdd(&acc5[0], log_z * log_z);
    atomicAdd(&acc5[1], m0);
    atomicAdd(&acc5[2], m1);
    atomicAdd(&acc5[3], p0);
    atomicAdd(&acc5[4], p1);
}

// ---------------------------------------------------------------------------
// Kernel 2: prefix sums over 8 experts + finalize scalar losses
// ---------------------------------------------------------------------------
__global__ void scan_finalize_kernel(const int* __restrict__ counts,
                                     int* __restrict__ offsets, int* __restrict__ cursor,
                                     const float* __restrict__ acc5,
                                     float* __restrict__ out_losses) {
    if (threadIdx.x == 0) {
        int run = 0;
        for (int e = 0; e < NEXP; ++e) { offsets[e] = run; run += counts[e]; }
        float invT = 1.0f / (float)TOKENS;
        out_losses[0] = acc5[0] * invT;                      // z_loss
        float tpe0 = acc5[1] * invT, tpe1 = acc5[2] * invT;
        float rp0  = acc5[3] * invT, rp1  = acc5[4] * invT;
        out_losses[1] = 2.0f * (tpe0 * rp0 + tpe1 * rp1);    // aux_loss
    }
    if (threadIdx.x < NEXP) cursor[threadIdx.x] = 0;
}

// ---------------------------------------------------------------------------
// Kernel 3: scatter tokens into per-expert contiguous lists
// ---------------------------------------------------------------------------
__global__ __launch_bounds__(NTHREADS)
void scatter_kernel(const int* __restrict__ idx, const float* __restrict__ rw,
                    const int* __restrict__ offsets, int* __restrict__ cursor,
                    int* __restrict__ tok_list, float* __restrict__ slot_w) {
    int t = blockIdx.x * NTHREADS + threadIdx.x;
#pragma unroll
    for (int s = 0; s < 2; ++s) {
        int e = idx[t * 2 + s];
        int pos = atomicAdd(&cursor[e], 1);
        int slot = offsets[e] + pos;
        tok_list[slot] = t;
        slot_w[slot] = rw[t * 2 + s];
    }
}

// ---------------------------------------------------------------------------
// Kernel 4: fused expert FFN via WMMA.  One block = (expert, 48-token tile).
//   async-stage x -> f16 A-frags in LDS; up-proj (K=1024) -> SiLU -> LDS
//   -> down-proj (K=256/chunk) -> atomic scatter-add with routing weights
// ---------------------------------------------------------------------------
__global__ __launch_bounds__(NT_FFN)
void moe_ffn_kernel(const float* __restrict__ x,
                    const float* __restrict__ w1, const float* __restrict__ b1,
                    const float* __restrict__ w2, const float* __restrict__ b2,
                    const int* __restrict__ counts, const int* __restrict__ offsets,
                    const int* __restrict__ tok_list, const float* __restrict__ slot_w,
                    float* __restrict__ out) {
    constexpr int TILES_PER_E = (TOKENS + TILE_M - 1) / TILE_M;   // 86
    int e    = blockIdx.x / TILES_PER_E;
    int tile = blockIdx.x % TILES_PER_E;
    int cnt  = counts[e];
    if (tile * TILE_M >= cnt) return;                // uniform per block
    int base = offsets[e] + tile * TILE_M;
    int rem  = cnt - tile * TILE_M;
    int n    = rem < TILE_M ? rem : TILE_M;

    // x tile in A-fragment layout: [msub][kblock][lane][elem] f16  (96 KB)
    __shared__ _Float16 xaf[MSUB][HDIM / 32][32][16];
    // union region (24 KB): f32 x staging chunk  OR  SiLU(h1) A-fragments
    __shared__ __align__(16) char ubuf[TILE_M * HCHUNK * 4];
    __shared__ int   stok[TILE_M];
    __shared__ float ssw[TILE_M];

    float*    stagep = (float*)ubuf;                          // [TILE_M][HCHUNK]
    _Float16* h1f    = (_Float16*)ubuf;  // [MSUB][FCHUNK/32][32][16]

    int tid  = threadIdx.x;
    int lane = tid & 31;
    int wave = tid >> 5;
    int lo   = lane & 15;
    int hi   = lane >> 4;

    if (tid < TILE_M) {
        if (tid < n) { stok[tid] = tok_list[base + tid]; ssw[tid] = slot_w[base + tid]; }
        else         { stok[tid] = 0;                    ssw[tid] = 0.0f; }
    }
    __syncthreads();

    // ---- stage gathered x rows (f32) chunk-wise, repack to f16 A-fragments
    for (int h0 = 0; h0 < HDIM; h0 += HCHUNK) {
        {
            constexpr int V4_PER_CHUNK = TILE_M * HCHUNK / 4;   // 1536
#pragma unroll
            for (int q = 0; q < V4_PER_CHUNK / NT_FFN; ++q) {   // 3 per thread
                int v4 = tid + q * NT_FFN;
                int r  = v4 >> 5;                                // HCHUNK/4 = 32 per row
                int c4 = (v4 & 31) * 4;
                const float* src = &x[(size_t)stok[r] * HDIM + h0 + c4];
                float*       dst = &stagep[r * HCHUNK + c4];
#if defined(USE_ASYNC_LDS)
                __builtin_amdgcn_global_load_async_to_lds_b128(
                    (GLOBAL_AS v4i*)src, (LDS_AS v4i*)dst, 0, 0);
#else
                *(float4*)dst = *(const float4*)src;
#endif
            }
#if defined(USE_ASYNC_LDS)
#if __has_builtin(__builtin_amdgcn_s_wait_asynccnt)
            __builtin_amdgcn_s_wait_asynccnt(0);
#else
            asm volatile("s_wait_asynccnt 0" ::: "memory");
#endif
#endif
        }
        __syncthreads();
        // repack this 48 x 128 f32 chunk into f16 A-fragment order
        for (int v = tid; v < MSUB * (HCHUNK / 32) * 32 * 16; v += NT_FFN) {
            int i   = v & 15;
            int l   = (v >> 4) & 31;
            int kbl = (v >> 9) & 3;
            int ms  = v >> 11;
            int lh  = l >> 4;
            int k   = kbl * 32 + a_frag_k(i, lh);
            int row = ms * 16 + (l & 15);
            xaf[ms][(h0 >> 5) + kbl][l][i] = (_Float16)stagep[row * HCHUNK + k];
        }
        __syncthreads();
    }

    // ---- down-proj accumulators: this wave owns output cols [wave*64, +64)
    int nbase = wave * 64;
    v8f acc[MSUB][4];
#pragma unroll
    for (int j = 0; j < 4; ++j) {
        float bias = b2[(size_t)e * HDIM + nbase + j * 16 + lo];
#pragma unroll
        for (int ms = 0; ms < MSUB; ++ms) {
            v8f c;
#pragma unroll
            for (int r = 0; r < 8; ++r) c[r] = bias;
            acc[ms][j] = c;
        }
    }

    for (int f0 = 0; f0 < FDIM; f0 += FCHUNK) {
        // ---- up-proj: this wave computes h1 cols [f0+wave*16, +16), all 3 subtiles
        int fc = f0 + wave * 16;
        v8f c1[MSUB];
        {
            float bias = b1[(size_t)e * FDIM + fc + lo];
#pragma unroll
            for (int ms = 0; ms < MSUB; ++ms)
#pragma unroll
                for (int r = 0; r < 8; ++r) c1[ms][r] = bias;
        }
#pragma unroll 2
        for (int kb = 0; kb < HDIM / 32; ++kb) {
            // B fragment: lanes<16 K=kb*32+{0..7,16..23}, lanes>=16 K=kb*32+{8..15,24..31}, N=lo
            v16h b;
            const float* wp = &w1[((size_t)e * HDIM + kb * 32 + hi * 8) * FDIM + fc + lo];
#pragma unroll
            for (int i = 0; i < 16; ++i) b[i] = (_Float16)wp[(size_t)(i + (i >= 8 ? 8 : 0)) * FDIM];
#pragma unroll
            for (int ms = 0; ms < MSUB; ++ms) {
                v16h a = *(const v16h*)&xaf[ms][kb][lane][0];
                c1[ms] = __builtin_amdgcn_wmma_f32_16x16x32_f16(
                    false, a, false, b, (short)0, c1[ms], false, false);
            }
        }

        __syncthreads();   // previous chunk's down-proj done reading h1f
        // SiLU + repack into A-fragment layout for down-proj
#pragma unroll
        for (int ms = 0; ms < MSUB; ++ms) {
#pragma unroll
            for (int r = 0; r < 8; ++r) {
                float v = c1[ms][r];
                float s = v / (1.0f + __expf(-v));
                int M  = r + hi * 8;               // row within 16-subtile
                int kk = wave * 16 + lo;           // k within 256-chunk
                int kb2 = kk >> 5, kloc = kk & 31;
                int l2, i2;
                a_frag_inv(M, kloc, l2, i2);
                h1f[(((ms * (FCHUNK / 32) + kb2) * 32) + l2) * 16 + i2] = (_Float16)s;
            }
        }
        __syncthreads();

        // ---- down-proj: acc[ms][j] += h1f[ms] @ w2[e][f0:f0+256, nbase+j*16..]
#pragma unroll
        for (int j = 0; j < 4; ++j) {
#pragma unroll
            for (int kb = 0; kb < FCHUNK / 32; ++kb) {
                v16h b;
                const float* wp = &w2[((size_t)e * FDIM + f0 + kb * 32 + hi * 8) * HDIM
                                      + nbase + j * 16 + lo];
#pragma unroll
                for (int i = 0; i < 16; ++i) b[i] = (_Float16)wp[(size_t)(i + (i >= 8 ? 8 : 0)) * HDIM];
#pragma unroll
                for (int ms = 0; ms < MSUB; ++ms) {
                    v16h a = *(const v16h*)
                        &h1f[(((ms * (FCHUNK / 32) + kb) * 32) + lane) * 16];
                    acc[ms][j] = __builtin_amdgcn_wmma_f32_16x16x32_f16(
                        false, a, false, b, (short)0, acc[ms][j], false, false);
                }
            }
        }
    }

    // ---- epilogue: final[t] += cw * (h1@w2 + b2)
#pragma unroll
    for (int ms = 0; ms < MSUB; ++ms) {
#pragma unroll
        for (int j = 0; j < 4; ++j) {
#pragma unroll
            for (int r = 0; r < 8; ++r) {
                int row = ms * 16 + r + hi * 8;
                if (row < n) {
                    float v = acc[ms][j][r] * ssw[row];
                    atomicAdd(&out[(size_t)stok[row] * HDIM + nbase + j * 16 + lo], v);
                }
            }
        }
    }
}

// ---------------------------------------------------------------------------
// Launcher
// ---------------------------------------------------------------------------
extern "C" void kernel_launch(void* const* d_in, const int* in_sizes, int n_in,
                              void* d_out, int out_size, void* d_ws, size_t ws_size,
                              hipStream_t stream) {
    (void)in_sizes; (void)n_in; (void)out_size; (void)ws_size;
    const float* x  = (const float*)d_in[0];   // [T,H]
    const float* gw = (const float*)d_in[1];   // [E,H]
    const float* w1 = (const float*)d_in[2];   // [E,H,F]
    const float* b1 = (const float*)d_in[3];   // [E,F]
    const float* w2 = (const float*)d_in[4];   // [E,F,H]
    const float* b2 = (const float*)d_in[5];   // [E,H]
    float* out = (float*)d_out;                // T*H final, then z_loss, aux_loss

    char* ws = (char*)d_ws;
    float* rw       = (float*)ws; ws += (size_t)TOKENS * 2 * sizeof(float);
    int*   idx      = (int*)ws;   ws += (size_t)TOKENS * 2 * sizeof(int);
    int*   tok_list = (int*)ws;   ws += (size_t)TOKENS * 2 * sizeof(int);
    float* slot_w   = (float*)ws; ws += (size_t)TOKENS * 2 * sizeof(float);
    int*   counts   = (int*)ws;   ws += 16 * sizeof(int);
    int*   offsets  = (int*)ws;   ws += 16 * sizeof(int);
    int*   cursor   = (int*)ws;   ws += 16 * sizeof(int);
    float* acc5     = (float*)ws; ws += 16 * sizeof(float);

    zero_kernel<<<1024, NTHREADS, 0, stream>>>(out, TOKENS * HDIM + 2, acc5, counts);
    router_kernel<<<TOKENS / NTHREADS, NTHREADS, 0, stream>>>(x, gw, rw, idx, counts, acc5);
    scan_finalize_kernel<<<1, 32, 0, stream>>>(counts, offsets, cursor, acc5,
                                               out + (size_t)TOKENS * HDIM);
    scatter_kernel<<<TOKENS / NTHREADS, NTHREADS, 0, stream>>>(idx, rw, offsets, cursor,
                                                               tok_list, slot_w);
    constexpr int TILES_PER_E = (TOKENS + TILE_M - 1) / TILE_M;
    moe_ffn_kernel<<<NEXP * TILES_PER_E, NT_FFN, 0, stream>>>(
        x, w1, b1, w2, b2, counts, offsets, tok_list, slot_w, out);
}
